// SmearExpert_53008486367831
// MI455X (gfx1250) — matrix operand order repaired and checked by reading, not compiled, not run on hardware
//
#include <hip/hip_runtime.h>
#include <math.h>

typedef __attribute__((ext_vector_type(16))) __bf16 v16bf;
typedef __attribute__((ext_vector_type(8)))  float  v8f;

#define T_TOK  4096   // B*S
#define Hdim   1024
#define Ddim   4096
#define Eexp   8
#define TILE_M 32     // tokens per block (2 WMMA M-tiles)
#define DC     256    // D columns per chunk (z chunk kept in LDS)

// ---------------- routing: top-2 + renorm + expert gather lists ----------------
__global__ void routing_kernel(const float* __restrict__ rw,
                               float* __restrict__ sparse_out,
                               int* __restrict__ counts,
                               int* __restrict__ lists,
                               float* __restrict__ wvals) {
  int t = blockIdx.x * blockDim.x + threadIdx.x;
  if (t >= T_TOK) return;
  float w[Eexp];
#pragma unroll
  for (int i = 0; i < Eexp; ++i) w[i] = rw[t * Eexp + i];
  int i1 = 0;
#pragma unroll
  for (int i = 1; i < Eexp; ++i) if (w[i] > w[i1]) i1 = i;    // first max (jax tie order)
  int i2 = (i1 == 0) ? 1 : 0;
#pragma unroll
  for (int i = 0; i < Eexp; ++i) {
    if (i == i1 || i == i2) continue;
    if (w[i] > w[i2]) i2 = i;
  }
  float s  = w[i1] + w[i2] + 1e-8f;
  float w1 = w[i1] / s, w2 = w[i2] / s;
#pragma unroll
  for (int i = 0; i < Eexp; ++i) sparse_out[t * Eexp + i] = 0.0f;
  sparse_out[t * Eexp + i1] = w1;
  sparse_out[t * Eexp + i2] = w2;
  int s1 = atomicAdd(&counts[i1], 1);
  lists[i1 * T_TOK + s1] = t; wvals[i1 * T_TOK + s1] = w1;
  int s2 = atomicAdd(&counts[i2], 1);
  lists[i2 * T_TOK + s2] = t; wvals[i2 * T_TOK + s2] = w2;
}

// Branchless exact-erf GELU (A&S 7.1.26, |err| ~ 1.5e-7), no EXEC divergence.
__device__ __forceinline__ float gelu_exact(float v) {
  float xa = fabsf(v) * 0.70710678118654752f;
  float t  = __builtin_amdgcn_rcpf(1.0f + 0.3275911f * xa);
  float p  = t * (0.254829592f +
             t * (-0.284496736f +
             t * (1.421413741f +
             t * (-1.453152027f +
             t * 1.061405429f))));
  float er = 1.0f - p * __expf(-xa * xa);
  er = copysignf(er, v);
  return 0.5f * v * (1.0f + er);
}

// Stage a 32(K) x 256(N) fp32 chunk into LDS in WMMA B-fragment order.
// B layout (16-bit, 32x16 per tile): lane = (n&15) + 16*((k>>3)&1), slot = (k&7) + 8*(k>>4).
// Row pairs (k even, k+1) share a lane and adjacent slots -> packed 2xbf16 store.
__device__ __forceinline__ void stage_b_chunk(const float* __restrict__ W, int ldb,
                                              __bf16 (*wf)[32][16], int tid) {
#pragma unroll
  for (int j = 0; j < 4; ++j) {
    int u  = tid + j * 256;          // 0..1023 float4-pair units
    int rp = u >> 6;                 // row pair 0..15
    int c4 = (u & 63) << 2;          // col base 0..252
    int k0 = rp * 2;
    const float4 lo4 = *(const float4*)(W + (size_t)k0 * ldb + c4);
    const float4 hi4 = *(const float4*)(W + (size_t)(k0 + 1) * ldb + c4);
    float lo[4] = {lo4.x, lo4.y, lo4.z, lo4.w};
    float hi[4] = {hi4.x, hi4.y, hi4.z, hi4.w};
#pragma unroll
    for (int q = 0; q < 4; ++q) {
      int n     = c4 + q;
      int ntile = n >> 4;
      int lane2 = (n & 15) + (((k0 >> 3) & 1) << 4);
      int slot  = (k0 & 7) + ((k0 >> 4) << 3);  // even
      union { __bf16 b[2]; unsigned int u32; } pk;
      pk.b[0] = (__bf16)lo[q];
      pk.b[1] = (__bf16)hi[q];
      *(unsigned int*)&wf[ntile][lane2][slot] = pk.u32;
    }
  }
}

// A-fragment swizzle for 16-bit 16x32 A tiles: given row r (0..15) and even k (0..31),
// write the (k, k+1) pair packed. h=(k>>3)&1; lane=r+16h; slot=2v+(k&1).
__device__ __forceinline__ void store_a_pair(__bf16 (*af)[16], int r, int k,
                                             float lo, float hi) {
  int h    = (k >> 3) & 1;
  int base = k - 8 * h;
  int vv   = (base < 8) ? (base >> 1) : (4 + ((base - 16) >> 1));
  union { __bf16 b[2]; unsigned int u32; } pk;
  pk.b[0] = (__bf16)lo;
  pk.b[1] = (__bf16)hi;
  *(unsigned int*)&af[r + 16 * h][2 * vv] = pk.u32;
}

// ---------------- fused per-expert tile kernel: down->GELU->up ----------------
__global__ __launch_bounds__(256) void smear_main(
    const float* __restrict__ x,
    const float* __restrict__ Wd,     // [E][H][D]
    const float* __restrict__ bd,     // [E][D]
    const float* __restrict__ Wu,     // [E][D][H]
    const int*   __restrict__ counts,
    const int*   __restrict__ lists,
    const float* __restrict__ wvals,
    float* __restrict__ out)          // [T][H], accumulated atomically
{
  const int e  = blockIdx.y;
  const int mt = blockIdx.x;
  const int cnt = counts[e];
  if (mt * TILE_M >= cnt) return;     // uniform per block -> EXEC stays all-ones

  // LDS, all operands pre-swizzled into WMMA fragment order (contiguous 32B/lane)
  __shared__ __align__(32) __bf16 xsf[2][32][32][16];   // 64KB: x tile A-frags
  __shared__ __align__(32) __bf16 zsf[2][8][32][16];    // 16KB: GELU(z) A-frags
  __shared__ __align__(32) __bf16 wfrag[2][16][32][16]; // 32KB: B-frags, ping-pong
  __shared__ int    trow[TILE_M];
  __shared__ float  wrow[TILE_M];

  const int tid = threadIdx.x;
  if (tid < TILE_M) {
    int r = mt * TILE_M + tid;
    if (r < cnt) { trow[tid] = lists[e * T_TOK + r]; wrow[tid] = wvals[e * T_TOK + r]; }
    else         { trow[tid] = lists[e * T_TOK + mt * TILE_M]; wrow[tid] = 0.0f; }
  }
  __syncthreads();

  // Load x tile (32 x 1024) as float4, convert+swizzle into A-fragment order.
#pragma unroll
  for (int j = 0; j < 32; ++j) {
    int i4 = tid + j * 256;            // 0..8191
    int r  = i4 >> 8;                  // 0..31
    int c4 = (i4 & 255) << 2;          // 0..1020
    const float4 v4 = *(const float4*)(x + (size_t)trow[r] * Hdim + c4);
    int m = r >> 4, rl = r & 15;
    int kc = c4 >> 5, kk = c4 & 31;    // kk multiple of 4
    store_a_pair(xsf[m][kc], rl, kk,     v4.x, v4.y);
    store_a_pair(xsf[m][kc], rl, kk + 2, v4.z, v4.w);
  }
  __syncthreads();

  const int wv   = tid >> 5;   // wave id 0..7
  const int lane = tid & 31;
  const int ln   = lane & 15;
  const int lh   = lane >> 4;

  const float* Wde = Wd + (size_t)e * Hdim * Ddim;
  const float* Wue = Wu + (size_t)e * Ddim * Hdim;
  const float* bde = bd + (size_t)e * Ddim;

  v8f oacc[4][2][2];                   // [H quarter][ntile half][m]
#pragma unroll
  for (int q = 0; q < 4; ++q)
#pragma unroll
    for (int t2 = 0; t2 < 2; ++t2)
#pragma unroll
      for (int m = 0; m < 2; ++m) oacc[q][t2][m] = (v8f){0,0,0,0,0,0,0,0};

  for (int dc = 0; dc < Ddim / DC; ++dc) {
    // ---- down-proj chunk: z[32 x DC] = GELU(x @ Wd[:, dc*DC..] + b) ----
    v8f dacc[2][2];
#pragma unroll
    for (int m = 0; m < 2; ++m)
#pragma unroll
      for (int t2 = 0; t2 < 2; ++t2) dacc[m][t2] = (v8f){0,0,0,0,0,0,0,0};

    for (int kc = 0; kc < Hdim / 32; ++kc) {
      stage_b_chunk(Wde + (size_t)(kc * 32) * Ddim + dc * DC, Ddim, wfrag[kc & 1], tid);
      __syncthreads();
#pragma unroll
      for (int t2 = 0; t2 < 2; ++t2) {
        v16bf b = *(const v16bf*)&wfrag[kc & 1][wv + 8 * t2][lane][0];
#pragma unroll
        for (int m = 0; m < 2; ++m) {
          v16bf a = *(const v16bf*)&xsf[m][kc][lane][0];
          dacc[m][t2] = __builtin_amdgcn_wmma_f32_16x16x32_bf16(
              false, a, false, b, (short)0, dacc[m][t2], false, false);
        }
      }
    }

    // bias + GELU, scatter into zsf in A-fragment order for the up-proj
#pragma unroll
    for (int t2 = 0; t2 < 2; ++t2) {
      int n0 = (wv + 8 * t2) * 16;
      float bias = bde[dc * DC + n0 + ln];
#pragma unroll
      for (int m = 0; m < 2; ++m) {
#pragma unroll
        for (int v = 0; v < 8; ++v) {
          float g = gelu_exact(dacc[m][t2][v] + bias);
          int Mloc = v + 8 * lh;            // 0..15
          int kcol = n0 + ln;               // 0..255 (K index for up-proj)
          int kc2  = kcol >> 5, kk = kcol & 31;
          int h2   = (kk >> 3) & 1;
          int base = kk - 8 * h2;
          int vv   = (base < 8) ? (base >> 1) : (4 + ((base - 16) >> 1));
          zsf[m][kc2][Mloc + 16 * h2][2 * vv + (kk & 1)] = (__bf16)g;
        }
      }
    }
    __syncthreads();

    // ---- up-proj partial: oacc += z(32 x DC) @ Wu[dc*DC.., :] ----
    for (int kc = 0; kc < DC / 32; ++kc) {
#pragma unroll
      for (int q = 0; q < 4; ++q) {
        int it = kc * 4 + q;
        stage_b_chunk(Wue + (size_t)(dc * DC + kc * 32) * Hdim + q * 256, Hdim,
                      wfrag[it & 1], tid);
        __syncthreads();
#pragma unroll
        for (int t2 = 0; t2 < 2; ++t2) {
          v16bf b = *(const v16bf*)&wfrag[it & 1][wv + 8 * t2][lane][0];
#pragma unroll
          for (int m = 0; m < 2; ++m) {
            v16bf a = *(const v16bf*)&zsf[m][kc][lane][0];
            oacc[q][t2][m] = __builtin_amdgcn_wmma_f32_16x16x32_bf16(
                false, a, false, b, (short)0, oacc[q][t2][m], false, false);
          }
        }
      }
    }
    __syncthreads();   // zsf reused next chunk
  }

  // ---- scale by routing weight, accumulate into out ----
#pragma unroll
  for (int q = 0; q < 4; ++q)
#pragma unroll
    for (int t2 = 0; t2 < 2; ++t2) {
      int cb = q * 256 + (wv + 8 * t2) * 16 + ln;
#pragma unroll
      for (int m = 0; m < 2; ++m)
#pragma unroll
        for (int v = 0; v < 8; ++v) {
          int Mg = m * 16 + v + 8 * lh;
          atomicAdd(&out[(size_t)trow[Mg] * Hdim + cb], oacc[q][t2][m][v] * wrow[Mg]);
        }
    }
}

extern "C" void kernel_launch(void* const* d_in, const int* in_sizes, int n_in,
                              void* d_out, int out_size, void* d_ws, size_t ws_size,
                              hipStream_t stream) {
  const float* x  = (const float*)d_in[0];
  const float* rw = (const float*)d_in[1];
  const float* Wd = (const float*)d_in[2];
  const float* bd = (const float*)d_in[3];
  const float* Wu = (const float*)d_in[4];
  (void)in_sizes; (void)n_in; (void)out_size; (void)ws_size;  // top_k fixed at 2

  float* out    = (float*)d_out;                  // [T][H]
  float* sparse = out + (size_t)T_TOK * Hdim;     // [T][E]

  int*   counts = (int*)d_ws;
  int*   lists  = counts + 8;                     // [E][T]
  float* wvals  = (float*)(lists + Eexp * T_TOK); // [E][T]

  hipMemsetAsync(out, 0, (size_t)T_TOK * Hdim * sizeof(float), stream);
  hipMemsetAsync(counts, 0, Eexp * sizeof(int), stream);

  routing_kernel<<<dim3(T_TOK / 256), dim3(256), 0, stream>>>(rw, sparse, counts, lists, wvals);

  dim3 grid(T_TOK / TILE_M, Eexp);   // worst case: all tokens on one expert
  smear_main<<<grid, dim3(256), 0, stream>>>(x, Wd, bd, Wu, counts, lists, wvals, out);
}
